// PitchAndDurationExtractor_51917564674246
// MI455X (gfx1250) — matrix operand order repaired and checked by reading, not compiled, not run on hardware
//
#include <hip/hip_runtime.h>
#include <math.h>
#include <stdint.h>

#define NEGV (-1e9f)

typedef __attribute__((ext_vector_type(16))) _Float16 v16h;
typedef __attribute__((ext_vector_type(8)))  _Float16 v8h;
typedef __attribute__((ext_vector_type(8)))  float    v8f;
typedef __attribute__((ext_vector_type(4)))  unsigned int u32x4;
typedef __attribute__((ext_vector_type(8)))  int          i32x8;
typedef __attribute__((ext_vector_type(4)))  int          i32x4;

namespace {
constexpr int BATCH = 32;
constexpr int TT    = 400;    // text frames
constexpr int TF    = 2000;   // mel frames
constexpr int HID   = 256;
constexpr int NMEL  = 80;
constexpr int NMELP = 96;     // padded to multiple of 32
}

#if __has_builtin(__builtin_amdgcn_tensor_load_to_lds)
#define HAVE_TDM 1
#else
#define HAVE_TDM 0
#endif

// ---------------------------------------------------------------------------
// Fragment loaders following CDNA5 ISA 7.12.2 wave32 layouts (16-bit WMMA).
// A (16x32, MxK): lane l holds row m=l&15; elements e=0..7 -> K = 8*(l>>4)+e,
//                 e=8..15 -> K = 16 + 8*(l>>4) + (e-8).  => two 16B chunks.
// B (32x16, KxN): lane l holds col n=l&15; same K mapping as A:
//                 e=0..7 -> K = 8*(l>>4)+e, e=8..15 -> K = 16 + 8*(l>>4) + (e-8).
// C/D (16x16 f32): VGPR r, lane l -> D[m=r+8*(l>>4)][n=l&15].
// ---------------------------------------------------------------------------
static __device__ inline v16h load_a16x32(const _Float16* rowk0, int hi) {
  v8h c0 = *reinterpret_cast<const v8h*>(rowk0 + hi * 8);
  v8h c1 = *reinterpret_cast<const v8h*>(rowk0 + 16 + hi * 8);
  v16h a;
#pragma unroll
  for (int e = 0; e < 8; ++e) { a[e] = c0[e]; a[e + 8] = c1[e]; }
  return a;
}

static __device__ inline v16h zero_frag() {
  v16h a;
#pragma unroll
  for (int e = 0; e < 16; ++e) a[e] = (_Float16)0.0f;
  return a;
}

static __device__ inline v16h load_b32x16(const _Float16* rowk0, int hi) {
  v8h c0 = *reinterpret_cast<const v8h*>(rowk0 + hi * 8);
  v8h c1 = *reinterpret_cast<const v8h*>(rowk0 + 16 + hi * 8);
  v16h b;
#pragma unroll
  for (int e = 0; e < 8; ++e) { b[e] = c0[e]; b[e + 8] = c1[e]; }
  return b;
}

// ---------------------------------------------------------------------------
// Tensor Data Mover: DMA a 2D tile (tileRows x rowElems f16, row stride
// strideElems) from global into LDS. Rows >= tensorRows zero-fill (ISA OOB
// rule). Descriptor packed per CDNA5 ISA 8.3/8.4; issued by one wave,
// completion via TENSORcnt.
// ---------------------------------------------------------------------------
#if HAVE_TDM
static __device__ inline void tdm_load_2d(void* lds_dst, const void* gsrc,
                                          unsigned rowElems, unsigned tileRows,
                                          unsigned tensorRows,
                                          unsigned strideElems) {
  unsigned long long ga = (unsigned long long)(uintptr_t)gsrc;
  unsigned lds = (unsigned)(uintptr_t)lds_dst;   // low 32b = LDS offset
  u32x4 g0;
  g0[0] = 1u;                                        // count=1, user mode
  g0[1] = lds;                                       // lds_addr
  g0[2] = (unsigned)(ga & 0xFFFFFFFFu);              // global_addr[31:0]
  g0[3] = (unsigned)((ga >> 32) & 0x01FFFFFFu) | (2u << 30);  // addr hi | type=2
  i32x8 g1;
  g1[0] = 0x00010000;                                // data_size=1 (2 bytes)
  g1[1] = (int)((rowElems & 0xFFFFu) << 16);         // tensor_dim0[15:0]
  g1[2] = (int)(((rowElems >> 16) & 0xFFFFu) |
                ((tensorRows & 0xFFFFu) << 16));     // dim0 hi | tensor_dim1 lo
  g1[3] = (int)(((tensorRows >> 16) & 0xFFFFu) |
                ((rowElems & 0xFFFFu) << 16));       // dim1 hi | tile_dim0
  g1[4] = (int)(tileRows & 0xFFFFu);                 // tile_dim1 | tile_dim2=0
  g1[5] = (int)strideElems;                          // tensor_dim0_stride lo
  g1[6] = 0;
  g1[7] = 0;
  i32x4 g2 = {0, 0, 0, 0};
  i32x4 g3 = {0, 0, 0, 0};
#if defined(__clang_major__) && __clang_major__ >= 23
  i32x8 g4 = {0, 0, 0, 0, 0, 0, 0, 0};
  __builtin_amdgcn_tensor_load_to_lds(g0, g1, g2, g3, g4, 0);
#else
  __builtin_amdgcn_tensor_load_to_lds(g0, g1, g2, g3, 0);
#endif
  __builtin_amdgcn_s_wait_tensorcnt(0);
}
#endif

// Fallback cooperative copy (block-wide), only used if TDM builtin missing.
static __device__ inline void coop_copy_2d(_Float16* dst, const _Float16* src,
                                           unsigned rowElems, unsigned tileRows,
                                           unsigned tensorRows,
                                           unsigned strideElems) {
  const unsigned total = tileRows * rowElems;
  for (unsigned i = threadIdx.x; i < total; i += blockDim.x) {
    unsigned r = i / rowElems, c = i - r * rowElems;
    dst[i] = (r < tensorRows) ? src[(size_t)r * strideElems + c]
                              : (_Float16)0.0f;
  }
}

// ---------------------------------------------------------------------------
// Prep kernels
// ---------------------------------------------------------------------------
__global__ void k_cast_f16(const float* __restrict__ src,
                           _Float16* __restrict__ dst, int n) {
  int i = blockIdx.x * blockDim.x + threadIdx.x;
  if (i < n) dst[i] = (_Float16)src[i];
}

__global__ void k_pad_mels(const float* __restrict__ mels,
                           _Float16* __restrict__ dst, int n) {
  int i = blockIdx.x * blockDim.x + threadIdx.x;   // over BATCH*TF*NMELP
  if (i >= n) return;
  int c  = i % NMELP;
  int bt = i / NMELP;                               // b*TF + t
  float v = (c < NMEL) ? mels[(long)bt * NMEL + c] : 0.0f;
  dst[i] = (_Float16)v;
}

// (Cout, Cin, 3) f32 -> (Cout, 3*Cpad) f16, K index = kk*Cpad + ci, zero pad
__global__ void k_pack3(const float* __restrict__ src, _Float16* __restrict__ dst,
                        int Cin, int Cpad, int n) {
  int i = blockIdx.x * blockDim.x + threadIdx.x;   // over 256*3*Cpad
  if (i >= n) return;
  int K  = 3 * Cpad;
  int co = i / K;
  int r  = i - co * K;
  int kk = r / Cpad;
  int ci = r - kk * Cpad;
  float v = (ci < Cin) ? src[((long)co * Cin + ci) * 3 + kk] : 0.0f;
  dst[i] = (_Float16)v;
}

__global__ void k_zero_bin(float* __restrict__ p) { *p = 0.0f; }

// ---------------------------------------------------------------------------
// Conv-as-GEMM with WMMA. out[b,t,co] = act( sum_{kk,ci} in[b,t+kk-pad,ci] *
// W[co, kk*CPAD+ci] + bias[co] ).  Block = 4 waves x (16 rows each) x 64 cols.
// Per tap: TDM stages the 64xCPAD weight tile into LDS (shared by all waves),
// then the fully-unrolled c0 loop streams A from global and B from LDS.
// ---------------------------------------------------------------------------
template <int TAPS, int CPAD, bool RELU>
__global__ __launch_bounds__(128)
void k_conv_gemm(const _Float16* __restrict__ in,   // (BATCH, T, CPAD)
                 const _Float16* __restrict__ W,    // (256, TAPS*CPAD) K-contig
                 const float*    __restrict__ bias, // (256)
                 _Float16*       __restrict__ out,  // (BATCH, T, 256)
                 int T) {
  constexpr int KDIM = TAPS * CPAD;
  constexpr int PAD  = (TAPS - 1) / 2;
  __shared__ __align__(16) _Float16 smem[64 * CPAD];

  const int lane = threadIdx.x & 31;
  const int wave = threadIdx.x >> 5;
  const int m0 = (blockIdx.x * 4 + wave) * 16;      // may exceed T: guarded
  const int n0 = blockIdx.y * 64;
  const int b  = blockIdx.z;
  const int hi = lane >> 4, ln = lane & 15;
  const _Float16* inB = in + (size_t)b * T * CPAD;

  v8f acc[4];
#pragma unroll
  for (int j = 0; j < 4; ++j)
#pragma unroll
    for (int r = 0; r < 8; ++r) acc[j][r] = 0.0f;

  const int row = m0 + ln;
#pragma unroll
  for (int kk = 0; kk < TAPS; ++kk) {
    if (kk) __syncthreads();                        // previous tile consumed
#if HAVE_TDM
    if (wave == 0)
      tdm_load_2d(smem, W + (size_t)n0 * KDIM + kk * CPAD,
                  (unsigned)CPAD, 64u, 64u, (unsigned)KDIM);
#else
    coop_copy_2d(smem, W + (size_t)n0 * KDIM + kk * CPAD,
                 (unsigned)CPAD, 64u, 64u, (unsigned)KDIM);
#endif
    __syncthreads();

    const int tt = row + kk - PAD;
    const bool av = (tt >= 0) && (tt < T);
    const _Float16* ap = inB + (size_t)tt * CPAD;
#pragma unroll
    for (int c0 = 0; c0 < CPAD; c0 += 32) {
      v16h a = av ? load_a16x32(ap + c0, hi) : zero_frag();
#pragma unroll
      for (int j = 0; j < 4; ++j) {
        v16h bf = load_b32x16(&smem[(size_t)(j * 16 + ln) * CPAD + c0], hi);
        acc[j] = __builtin_amdgcn_wmma_f32_16x16x32_f16(
            false, a, false, bf, (short)0, acc[j], false, false);
      }
    }
  }

  _Float16* outB = out + (size_t)b * T * HID;
#pragma unroll
  for (int j = 0; j < 4; ++j) {
    const int n = n0 + j * 16 + ln;
    const float bv = bias[n];
#pragma unroll
    for (int r = 0; r < 8; ++r) {
      const int m = m0 + r + hi * 8;
      float v = acc[j][r] + bv;
      if (RELU) v = fmaxf(v, 0.0f);
      if (m < T) outB[(size_t)m * HID + n] = (_Float16)v;
    }
  }
}

// ---------------------------------------------------------------------------
// Row sum-of-squares over H=256 (vectorized 16B loads)
// ---------------------------------------------------------------------------
__global__ void k_rowsq(const _Float16* __restrict__ x,
                        float* __restrict__ sq, int rows) {
  int r = blockIdx.x * blockDim.x + threadIdx.x;
  if (r >= rows) return;
  const v8h* p = reinterpret_cast<const v8h*>(x + (size_t)r * HID);
  float s = 0.0f;
  for (int h = 0; h < HID / 8; ++h) {
    v8h v = p[h];
#pragma unroll
    for (int e = 0; e < 8; ++e) { float f = (float)v[e]; s += f * f; }
  }
  sq[r] = s;
}

// ---------------------------------------------------------------------------
// Score: G = featF . textF^T (K=256) via WMMA, fused -sqrt(||f-t||^2) + mask.
// textF tile (64 rows, zero-filled past TT via TDM OOB) staged in LDS once.
// Writes raw masked score into log_p region of d_out (log-softmax later).
// ---------------------------------------------------------------------------
__global__ __launch_bounds__(128)
void k_score(const _Float16* __restrict__ featF,  // (BATCH, TF, 256)
             const _Float16* __restrict__ textF,  // (BATCH, TT, 256)
             const float*    __restrict__ fsq,    // (BATCH*TF)
             const float*    __restrict__ tsq,    // (BATCH*TT)
             const int*      __restrict__ plens,  // (BATCH)
             float*          __restrict__ logp) { // (BATCH, TF, TT)
  __shared__ __align__(16) _Float16 smem[64 * HID];
  const int lane = threadIdx.x & 31;
  const int wave = threadIdx.x >> 5;
  const int m0 = (blockIdx.x * 4 + wave) * 16;     // may exceed TF: guarded
  const int n0 = blockIdx.y * 64;
  const int b  = blockIdx.z;
  const int hi = lane >> 4, ln = lane & 15;
  const _Float16* fB = featF + (size_t)b * TF * HID;
  const _Float16* tB = textF + (size_t)b * TT * HID;

  const unsigned tRows = (unsigned)(TT - n0);       // n0 < TT always
#if HAVE_TDM
  if (wave == 0)
    tdm_load_2d(smem, tB + (size_t)n0 * HID, (unsigned)HID, 64u, tRows,
                (unsigned)HID);
#else
  coop_copy_2d(smem, tB + (size_t)n0 * HID, (unsigned)HID, 64u, tRows,
               (unsigned)HID);
#endif
  __syncthreads();

  v8f acc[4];
#pragma unroll
  for (int j = 0; j < 4; ++j)
#pragma unroll
    for (int r = 0; r < 8; ++r) acc[j][r] = 0.0f;

  const int fr = m0 + ln;
  const bool av = fr < TF;
  const _Float16* ap = fB + (size_t)fr * HID;
#pragma unroll
  for (int k0 = 0; k0 < HID; k0 += 32) {
    v16h a = av ? load_a16x32(ap + k0, hi) : zero_frag();
#pragma unroll
    for (int j = 0; j < 4; ++j) {
      v16h bf = load_b32x16(&smem[(size_t)(j * 16 + ln) * HID + k0], hi);
      acc[j] = __builtin_amdgcn_wmma_f32_16x16x32_f16(
          false, a, false, bf, (short)0, acc[j], false, false);
    }
  }

  const int plen = plens[b];
  float* lpB = logp + (size_t)b * TF * TT;
#pragma unroll
  for (int j = 0; j < 4; ++j) {
    const int n = n0 + j * 16 + ln;
    if (n >= TT) continue;
    const float tq = tsq[(size_t)b * TT + n];
#pragma unroll
    for (int r = 0; r < 8; ++r) {
      const int m = m0 + r + hi * 8;
      if (m >= TF) continue;
      float sqv = fsq[(size_t)b * TF + m] + tq - 2.0f * acc[j][r];
      float sc = -sqrtf(fmaxf(sqv, 1e-12f));
      if (n >= plen) sc = NEGV;
      lpB[(size_t)m * TT + n] = sc;
    }
  }
}

// ---------------------------------------------------------------------------
// In-place log-softmax over last axis (length TT); one block per (b,f) row.
// ---------------------------------------------------------------------------
__global__ __launch_bounds__(128)
void k_logsoftmax(float* __restrict__ logp) {
  __shared__ float red[128];
  float* row = logp + (size_t)blockIdx.x * TT;
  const int t = threadIdx.x;

  float m = -3.4e38f;
  for (int i = t; i < TT; i += 128) m = fmaxf(m, row[i]);
  red[t] = m; __syncthreads();
  for (int s = 64; s > 0; s >>= 1) {
    if (t < s) red[t] = fmaxf(red[t], red[t + s]);
    __syncthreads();
  }
  m = red[0]; __syncthreads();

  float s = 0.0f;
  for (int i = t; i < TT; i += 128) s += expf(row[i] - m);
  red[t] = s; __syncthreads();
  for (int st = 64; st > 0; st >>= 1) {
    if (t < st) red[t] += red[t + st];
    __syncthreads();
  }
  const float lse = m + logf(red[0]);
  for (int i = t; i < TT; i += 128) row[i] = row[i] - lse;
}

// ---------------------------------------------------------------------------
// Monotonic alignment search: one block per batch element.
// Forward DP: 400 states in parallel, ping-pong LDS Q. Backtrack: thread 0.
// ---------------------------------------------------------------------------
__global__ __launch_bounds__(512)
void k_mas(const float* __restrict__ logp,    // (BATCH, TF, TT), log-softmaxed
           const float* __restrict__ pitch,   // (BATCH, TF, 1)
           const int*   __restrict__ mlens,
           const int*   __restrict__ plens,
           unsigned char* __restrict__ takes, // (BATCH, TF, TT)
           float* __restrict__ out_avg,       // (BATCH, TT)
           float* __restrict__ out_ds,        // (BATCH, TT)
           float* __restrict__ out_bin) {
  __shared__ float Qa[TT], Qb[TT], dsArr[TT], psArr[TT];
  const int b   = blockIdx.x;
  const int tid = threadIdx.x;
  const int ml  = mlens[b];
  const int pl  = plens[b];
  const float* lpB = logp + (size_t)b * TF * TT;
  unsigned char* tkB = takes + (size_t)b * TF * TT;

  float* Qc = Qa;
  float* Qn = Qb;
  if (tid < TT) Qc[tid] = NEGV;
  __syncthreads();

  for (int j = 0; j < ml; ++j) {
    if (tid < TT) {
      const float lp = lpB[(size_t)j * TT + tid];
      const float qi = Qc[tid];
      const float sh = (tid > 0) ? Qc[tid - 1] : NEGV;
      const bool take = (sh > qi) && (j > 0);
      float qn;
      if (j == 0) qn = (tid == 0) ? lp : NEGV;
      else        qn = lp + fmaxf(sh, qi);
      tkB[(size_t)j * TT + tid] = take ? 1 : 0;
      Qn[tid] = qn;
    }
    __syncthreads();
    float* tmp = Qc; Qc = Qn; Qn = tmp;
  }

  if (tid < TT) { dsArr[tid] = 0.0f; psArr[tid] = 0.0f; }
  __syncthreads();

  if (tid == 0) {
    int i = 0;
    float lsum = 0.0f;
    for (int j = TF - 1; j >= 0; --j) {
      if (j == ml - 1) i = pl - 1;
      const int a = i;
      int move = 0;
      if (j >= 1 && j <= ml - 1) move = (int)tkB[(size_t)j * TT + a];
      if (j < ml) {
        dsArr[a] += 1.0f;
        lsum += lpB[(size_t)j * TT + a];
        psArr[a] += pitch[(size_t)b * TF + j];
      }
      i = a - move;
    }
    atomicAdd(out_bin, (-lsum / (float)ml) / (float)BATCH);
  }
  __syncthreads();

  if (tid < TT) {
    const float d = dsArr[tid];
    out_ds[(size_t)b * TT + tid]  = d;
    out_avg[(size_t)b * TT + tid] = (d > 0.0f) ? (psArr[tid] / fmaxf(d, 1.0f)) : 0.0f;
  }
}

// ---------------------------------------------------------------------------
extern "C" void kernel_launch(void* const* d_in, const int* in_sizes, int n_in,
                              void* d_out, int out_size, void* d_ws, size_t ws_size,
                              hipStream_t stream) {
  (void)in_sizes; (void)n_in; (void)out_size; (void)ws_size;

  const float* hs      = (const float*)d_in[0];
  const float* pitches = (const float*)d_in[1];
  const float* mels    = (const float*)d_in[2];
  const int*   plens   = (const int*)d_in[3];
  const int*   mlens   = (const int*)d_in[4];
  const float* tw1 = (const float*)d_in[5];  const float* tb1 = (const float*)d_in[6];
  const float* tw2 = (const float*)d_in[7];  const float* tb2 = (const float*)d_in[8];
  const float* fw1 = (const float*)d_in[9];  const float* fb1 = (const float*)d_in[10];
  const float* fw2 = (const float*)d_in[11]; const float* fb2 = (const float*)d_in[12];
  const float* fw3 = (const float*)d_in[13]; const float* fb3 = (const float*)d_in[14];

  // d_out layout: avg (32*400) | ds (32*400) | log_p_attn (32*2000*400) | bin (1)
  float* out      = (float*)d_out;
  float* out_avg  = out;
  float* out_ds   = out + BATCH * TT;
  float* logp     = out + 2 * BATCH * TT;
  float* out_bin  = logp + (size_t)BATCH * TF * TT;

  // Workspace carve (256B aligned slices)
  char* w = (char*)d_ws;
  auto alloc = [&](size_t bytes) -> void* {
    void* p = (void*)w;
    w += (bytes + 255) & ~((size_t)255);
    return p;
  };
  _Float16* hsH      = (_Float16*)alloc((size_t)BATCH * TT * HID * 2);
  _Float16* melsP    = (_Float16*)alloc((size_t)BATCH * TF * NMELP * 2);
  _Float16* Wt1      = (_Float16*)alloc((size_t)HID * 3 * HID * 2);
  _Float16* Wt2      = (_Float16*)alloc((size_t)HID * HID * 2);
  _Float16* Wf1      = (_Float16*)alloc((size_t)HID * 3 * NMELP * 2);
  _Float16* Wf2      = (_Float16*)alloc((size_t)HID * 3 * HID * 2);
  _Float16* Wf3      = (_Float16*)alloc((size_t)HID * HID * 2);
  _Float16* text_mid = (_Float16*)alloc((size_t)BATCH * TT * HID * 2);
  _Float16* textF    = (_Float16*)alloc((size_t)BATCH * TT * HID * 2);
  _Float16* f1       = (_Float16*)alloc((size_t)BATCH * TF * HID * 2);
  _Float16* f2       = (_Float16*)alloc((size_t)BATCH * TF * HID * 2);
  _Float16* featF    = (_Float16*)alloc((size_t)BATCH * TF * HID * 2);
  float*    tsq      = (float*)alloc((size_t)BATCH * TT * 4);
  float*    fsq      = (float*)alloc((size_t)BATCH * TF * 4);
  unsigned char* tks = (unsigned char*)alloc((size_t)BATCH * TF * TT);

  const int THR = 256;
  auto nb = [](int n, int t) { return (n + t - 1) / t; };

  // --- prep ---
  k_cast_f16<<<nb(BATCH * TT * HID, THR), THR, 0, stream>>>(hs, hsH, BATCH * TT * HID);
  k_pad_mels<<<nb(BATCH * TF * NMELP, THR), THR, 0, stream>>>(mels, melsP, BATCH * TF * NMELP);
  k_pack3<<<nb(HID * 3 * HID, THR), THR, 0, stream>>>(tw1, Wt1, HID, HID, HID * 3 * HID);
  k_cast_f16<<<nb(HID * HID, THR), THR, 0, stream>>>(tw2, Wt2, HID * HID);
  k_pack3<<<nb(HID * 3 * NMELP, THR), THR, 0, stream>>>(fw1, Wf1, NMEL, NMELP, HID * 3 * NMELP);
  k_pack3<<<nb(HID * 3 * HID, THR), THR, 0, stream>>>(fw2, Wf2, HID, HID, HID * 3 * HID);
  k_cast_f16<<<nb(HID * HID, THR), THR, 0, stream>>>(fw3, Wf3, HID * HID);
  k_zero_bin<<<1, 1, 0, stream>>>(out_bin);

  // --- conv GEMMs (WMMA + TDM-staged weights) ---
  dim3 blk(128);
  dim3 gT(nb(TT / 16, 4), HID / 64, BATCH);   // (7,4,32)
  dim3 gF(nb(TF / 16, 4), HID / 64, BATCH);   // (32,4,32)
  k_conv_gemm<3, HID,   true ><<<gT, blk, 0, stream>>>(hsH,      Wt1, tb1, text_mid, TT);
  k_conv_gemm<1, HID,   false><<<gT, blk, 0, stream>>>(text_mid, Wt2, tb2, textF,    TT);
  k_conv_gemm<3, NMELP, true ><<<gF, blk, 0, stream>>>(melsP,    Wf1, fb1, f1,       TF);
  k_conv_gemm<3, HID,   true ><<<gF, blk, 0, stream>>>(f1,       Wf2, fb2, f2,       TF);
  k_conv_gemm<1, HID,   false><<<gF, blk, 0, stream>>>(f2,       Wf3, fb3, featF,    TF);

  // --- norms ---
  k_rowsq<<<nb(BATCH * TT, THR), THR, 0, stream>>>(textF, tsq, BATCH * TT);
  k_rowsq<<<nb(BATCH * TF, THR), THR, 0, stream>>>(featF, fsq, BATCH * TF);

  // --- score GEMM (WMMA + TDM-staged textF tile) fused distance + mask ---
  dim3 gS(nb(TF / 16, 4), nb(TT, 64), BATCH); // (32,7,32)
  k_score<<<gS, blk, 0, stream>>>(featF, textF, fsq, tsq, plens, logp);

  // --- log-softmax rows (in-place in d_out) ---
  k_logsoftmax<<<dim3(BATCH * TF), dim3(128), 0, stream>>>(logp);

  // --- MAS + reductions ---
  k_mas<<<dim3(BATCH), dim3(512), 0, stream>>>(logp, pitches, mlens, plens, tks,
                                               out_avg, out_ds, out_bin);
}
